// TransformerGraphEmbeddingCosine_43765716746927
// MI455X (gfx1250) — matrix operand order repaired and checked by reading, not compiled, not run on hardware
//
#include <hip/hip_runtime.h>
#include <math.h>
#include <stdint.h>

// ---------------- problem constants ----------------
#define N_NODES   10000
#define N_EDGES   160000
#define D_IN      8
#define D_MODEL   640
#define N_LAYERS  6
#define N_BATCH   64
#define LN_EPS    1e-5f
#define COS_EPS   1e-8f
#define INV_SQRT_D 0.0395284707521047f   // 1/sqrt(640)

#define BM 64          // block tile rows   (4 x 16)
#define BN 128         // block tile cols   (8 x 16)
#define KSTEP 32       // one WMMA K step
#define MP_NODE 10048  // N_NODES padded to multiple of BM

typedef __attribute__((ext_vector_type(16))) __bf16 v16bf;
typedef __attribute__((ext_vector_type(8)))  __bf16 v8bf;
typedef __attribute__((ext_vector_type(8)))  float  v8f;

// ---------------- helpers ----------------
__device__ __forceinline__ unsigned short f32_to_bf16_rne(float f) {
  unsigned int u = __float_as_uint(f);
  unsigned int r = u + 0x7FFFu + ((u >> 16) & 1u);
  if ((u & 0x7F800000u) == 0x7F800000u) r = u;   // inf/nan passthrough
  return (unsigned short)(r >> 16);
}

// order-preserving float->uint mapping (for atomicMax-based segment max)
__device__ __forceinline__ unsigned int f32_ord(float f) {
  unsigned int b = __float_as_uint(f);
  return b ^ ((unsigned int)((int)b >> 31) | 0x80000000u);
}
__device__ __forceinline__ float ord_f32(unsigned int k) {
  unsigned int b = k ^ ((k & 0x80000000u) ? 0x80000000u : 0xFFFFFFFFu);
  return __uint_as_float(b);
}

__device__ __forceinline__ float wave_reduce_add(float v) {
  #pragma unroll
  for (int off = 16; off > 0; off >>= 1) v += __shfl_xor(v, off, 32);
  return v;
}

// gfx1250 async global->LDS copy (ASYNCcnt-tracked); lds_off = low 32 bits of
// the generic shared pointer == byte offset within the workgroup LDS allocation.
__device__ __forceinline__ void async_ld_b128(unsigned lds_off, const void* gaddr) {
  asm volatile("global_load_async_to_lds_b128 %0, %1, off"
               :: "v"(lds_off), "v"((unsigned long long)(uintptr_t)gaddr)
               : "memory");
}

// ---------------- bf16 staging kernels ----------------
// src [rows_src x scols] f32 -> dst [rows_dst x dcols] bf16, zero pad rows/cols
__global__ void k_cvt_pad_bf16(const float* __restrict__ src, unsigned short* __restrict__ dst,
                               int rows_src, int rows_dst, int scols, int dcols) {
  long long i = (long long)blockIdx.x * blockDim.x + threadIdx.x;
  long long total = (long long)rows_dst * dcols;
  if (i >= total) return;
  int r = (int)(i / dcols), c = (int)(i % dcols);
  float v = (r < rows_src && c < scols) ? src[(long long)r * scols + c] : 0.0f;
  dst[i] = f32_to_bf16_rne(v);
}

// W [K x Ncols] f32 row-major -> Wt [Ncols x Kp] bf16 (transposed, K zero-padded to Kp)
__global__ void k_cvtT_pad_bf16(const float* __restrict__ W, unsigned short* __restrict__ Wt,
                                int K, int Ncols, int Kp) {
  long long i = (long long)blockIdx.x * blockDim.x + threadIdx.x;
  long long total = (long long)Ncols * Kp;
  if (i >= total) return;
  int n = (int)(i / Kp), k = (int)(i % Kp);
  float v = (k < K) ? W[(long long)k * Ncols + n] : 0.0f;
  Wt[i] = f32_to_bf16_rne(v);
}

// ---------------- WMMA bf16 GEMM with async-LDS double buffering ----------------
// C[M x Ncols] = A[Mpad x Kp] @ Bt[Ncols x Kp]^T + bias, optional ReLU.
// Block(256)=8 waves computes BM x BN; wave (wm, wng) computes M-tile wm and
// N-tiles wng*4..wng*4+3. A/B tiles staged in LDS via GLOBAL_LOAD_ASYNC_TO_LDS_B128,
// double-buffered, S_WAIT_ASYNCCNT for pipelining.
// WMMA fragment layouts used here (contiguous k order, same in A and B):
//   A 16x32: lane half h, row=lane&15; elems 0..15 = K(16h..16h+15)
//   B 32x16: col=lane&15; elems 0..15 = K(16h..16h+15)  (from transposed weights)
//   C 16x16: elem r -> row r+8h, col lane&15
__global__ void __launch_bounds__(256)
k_gemm_bf16_async(const __bf16* __restrict__ A, const __bf16* __restrict__ Bt,
                  const float* __restrict__ bias, float* __restrict__ C,
                  int M, int Ncols, int Kp, int relu) {
  __shared__ __align__(32) unsigned short Abuf[2][BM][KSTEP];   // 8 KB
  __shared__ __align__(32) unsigned short Bbuf[2][BN][KSTEP];   // 16 KB

  const int t = threadIdx.x;
  const int nBlk = Ncols / BN;
  const int bm = blockIdx.x / nBlk, bn = blockIdx.x % nBlk;
  const long long m0 = (long long)bm * BM;
  const int n0 = bn * BN;

  // stage-load addressing: thread t moves one A b128 and two B b128 per stage
  const int srow = t >> 2;              // 0..63
  const int skk  = (t & 3) * 8;         // 0,8,16,24
  const __bf16* gA  = A  + (m0 + srow) * Kp + skk;
  const __bf16* gB0 = Bt + (long long)(n0 + srow) * Kp + skk;
  const __bf16* gB1 = Bt + (long long)(n0 + 64 + srow) * Kp + skk;
  const unsigned ldsA  = (unsigned)(uintptr_t)&Abuf[0][srow][skk];
  const unsigned ldsB0 = (unsigned)(uintptr_t)&Bbuf[0][srow][skk];
  const unsigned ldsB1 = (unsigned)(uintptr_t)&Bbuf[0][64 + srow][skk];
  const unsigned aStride = (unsigned)sizeof(Abuf[0]);
  const unsigned bStride = (unsigned)sizeof(Bbuf[0]);

  const int lane = t & 31, wv = t >> 5;
  const int h = lane >> 4, l16 = lane & 15;
  const int wm = wv & 3;        // M-tile within block
  const int wng = wv >> 2;      // N-tile group (0/1) -> tiles wng*4 + 0..3

  v8f acc[4] = {};
  const int KT = Kp / KSTEP;

  // prologue: stage k-chunk 0 into buffer 0
  async_ld_b128(ldsA,  gA);
  async_ld_b128(ldsB0, gB0);
  async_ld_b128(ldsB1, gB1);

  for (int kt = 0; kt < KT; ++kt) {
    const int b = kt & 1;
    if (kt + 1 < KT) {
      const long long ko = (long long)(kt + 1) * KSTEP;
      const unsigned sel = (unsigned)((kt + 1) & 1);
      async_ld_b128(ldsA  + sel * aStride, gA  + ko);
      async_ld_b128(ldsB0 + sel * bStride, gB0 + ko);
      async_ld_b128(ldsB1 + sel * bStride, gB1 + ko);
      asm volatile("s_wait_asynccnt 0x3" ::: "memory");  // current stage landed
    } else {
      asm volatile("s_wait_asynccnt 0x0" ::: "memory");
    }
    __syncthreads();

    // A fragment (shared by this wave's 4 WMMAs)
    const unsigned short* ap = &Abuf[b][wm * 16 + l16][h * 16];
    v8bf alo = *(const v8bf*)ap;
    v8bf ahi = *(const v8bf*)(ap + 8);
    v16bf a;
    #pragma unroll
    for (int i = 0; i < 8; ++i) { a[i] = alo[i]; a[i + 8] = ahi[i]; }

    #pragma unroll
    for (int j = 0; j < 4; ++j) {
      const unsigned short* bp = &Bbuf[b][(wng * 4 + j) * 16 + l16][h * 16];
      v16bf bb = *(const v16bf*)bp;
      acc[j] = __builtin_amdgcn_wmma_f32_16x16x32_bf16(false, a, false, bb,
                                                       (short)0, acc[j], false, false);
    }
    __syncthreads();  // all reads of buffer b done before it is re-staged
  }

  // epilogue: bias (+ReLU) and guarded store
  const long long row0 = m0 + wm * 16 + 8 * h;
  #pragma unroll
  for (int j = 0; j < 4; ++j) {
    const int col = n0 + (wng * 4 + j) * 16 + l16;
    const float bv = bias ? bias[col] : 0.0f;
    #pragma unroll
    for (int r = 0; r < 8; ++r) {
      const long long row = row0 + r;
      if (row < M) {
        float v = acc[j][r] + bv;
        if (relu) v = fmaxf(v, 0.0f);
        C[row * Ncols + col] = v;
      }
    }
  }
}

// ---------------- per-node scalars + per-node state init ----------------
__global__ void k_node_scalars(const float* __restrict__ Q, const float* __restrict__ We,
                               const float* __restrict__ be,
                               float* __restrict__ qWe, float* __restrict__ qbe,
                               unsigned int* __restrict__ mkey, float* __restrict__ den,
                               float* __restrict__ sWa) {
  int node = blockIdx.x * (blockDim.x >> 5) + (threadIdx.x >> 5);
  if (node >= N_NODES) return;
  int lane = threadIdx.x & 31;
  const float* q = Q + (long long)node * D_MODEL;
  float s1 = 0.f, s2 = 0.f;
  for (int d = lane; d < D_MODEL; d += 32) { float qv = q[d]; s1 += qv * We[d]; s2 += qv * be[d]; }
  s1 = wave_reduce_add(s1); s2 = wave_reduce_add(s2);
  if (lane == 0) {
    qWe[node] = s1; qbe[node] = s2;
    mkey[node] = 0u;          // ordered-encoding -inf sentinel
    den[node] = 0.0f; sWa[node] = 0.0f;
  }
}

// ---------------- edge phase ----------------
__global__ void k_edge_logits(const float* __restrict__ Q, const float* __restrict__ Kf,
                              const int* __restrict__ eidx, const float* __restrict__ attr,
                              const float* __restrict__ qWe, const float* __restrict__ qbe,
                              float* __restrict__ logits, unsigned int* __restrict__ mkey) {
  int e = blockIdx.x * (blockDim.x >> 5) + (threadIdx.x >> 5);
  if (e >= N_EDGES) return;
  int lane = threadIdx.x & 31;
  int src = eidx[e], dst = eidx[N_EDGES + e];
  const float* q = Q  + (long long)dst * D_MODEL;
  const float* k = Kf + (long long)src * D_MODEL;
  float s = 0.f;
  for (int d = lane; d < D_MODEL; d += 32) s += q[d] * k[d];
  s = wave_reduce_add(s);
  if (lane == 0) {
    float lg = (s + attr[e] * qWe[dst] + qbe[dst]) * INV_SQRT_D;
    logits[e] = lg;
    atomicMax(&mkey[dst], f32_ord(lg));
  }
}

__global__ void k_edge_expsum(float* __restrict__ logits, const unsigned int* __restrict__ mkey,
                              const int* __restrict__ eidx, float* __restrict__ den) {
  int e = blockIdx.x * blockDim.x + threadIdx.x;
  if (e >= N_EDGES) return;
  int dst = eidx[N_EDGES + e];
  float ex = __expf(logits[e] - ord_f32(mkey[dst]));
  logits[e] = ex;
  atomicAdd(&den[dst], ex);
}

__global__ void k_edge_scatter(const float* __restrict__ ex, const float* __restrict__ den,
                               const int* __restrict__ eidx, const float* __restrict__ attr,
                               const float* __restrict__ V, float* __restrict__ XN,
                               float* __restrict__ sWa) {
  int e = blockIdx.x * (blockDim.x >> 5) + (threadIdx.x >> 5);
  if (e >= N_EDGES) return;
  int lane = threadIdx.x & 31;
  int src = eidx[e], dst = eidx[N_EDGES + e];
  float w = ex[e] / den[dst];
  if (lane == 0) atomicAdd(&sWa[dst], w * attr[e]);
  const float* v = V + (long long)src * D_MODEL;
  float* o = XN + (long long)dst * D_MODEL;
  for (int d = lane; d < D_MODEL; d += 32) atomicAdd(&o[d], w * v[d]);
}

// x_next = (x@Ws+bs + sum w*v) + sWa*We + be   (rank-1 edge-embedding closure)
__global__ void k_finalize(const float* __restrict__ XN, const float* __restrict__ sWa,
                           const float* __restrict__ We, const float* __restrict__ be,
                           float* __restrict__ X) {
  long long i = (long long)blockIdx.x * blockDim.x + threadIdx.x;
  if (i >= (long long)N_NODES * D_MODEL) return;
  int n = (int)(i / D_MODEL), d = (int)(i % D_MODEL);
  X[i] = XN[i] + sWa[n] * We[d] + be[d];
}

// ---------------- pooling / layernorm / cosine ----------------
__global__ void k_zero_f32(float* __restrict__ p, int count) {
  int i = blockIdx.x * blockDim.x + threadIdx.x;
  if (i < count) p[i] = 0.0f;
}

__global__ void k_pool(const float* __restrict__ X, const int* __restrict__ batch,
                       float* __restrict__ pooled) {
  long long i = (long long)blockIdx.x * blockDim.x + threadIdx.x;
  if (i >= (long long)N_NODES * D_MODEL) return;
  int n = (int)(i / D_MODEL), d = (int)(i % D_MODEL);
  atomicAdd(&pooled[(long long)batch[n] * D_MODEL + d], X[i]);
}

__global__ void k_layernorm_bf16(const float* __restrict__ pooled, const float* __restrict__ lnw,
                                 const float* __restrict__ lnb, unsigned short* __restrict__ Hbf) {
  int b = blockIdx.x;
  const float* p = pooled + (long long)b * D_MODEL;
  float s = 0.f, ss = 0.f;
  for (int d = threadIdx.x; d < D_MODEL; d += blockDim.x) { float v = p[d]; s += v; ss += v * v; }
  __shared__ float rs[8], rss[8];
  s = wave_reduce_add(s); ss = wave_reduce_add(ss);
  int lane = threadIdx.x & 31, wid = threadIdx.x >> 5;
  if (lane == 0) { rs[wid] = s; rss[wid] = ss; }
  __syncthreads();
  if (threadIdx.x == 0) {
    float a = 0.f, c = 0.f;
    for (int w = 0; w < (int)(blockDim.x >> 5); ++w) { a += rs[w]; c += rss[w]; }
    rs[0] = a; rss[0] = c;
  }
  __syncthreads();
  float mu   = rs[0] * (1.0f / D_MODEL);
  float var  = rss[0] * (1.0f / D_MODEL) - mu * mu;
  float rsig = rsqrtf(var + LN_EPS);
  for (int d = threadIdx.x; d < D_MODEL; d += blockDim.x) {
    float h = (p[d] - mu) * rsig * lnw[d] + lnb[d];
    Hbf[(long long)b * D_MODEL + d] = f32_to_bf16_rne(h);
  }
}

__global__ void k_cosine(const float* __restrict__ A, const float* __restrict__ Bm,
                         float* __restrict__ out) {
  int b = blockIdx.x;
  const float* pa = A  + (long long)b * D_MODEL;
  const float* pb = Bm + (long long)b * D_MODEL;
  float d = 0.f, na = 0.f, nb = 0.f;
  for (int i = threadIdx.x; i < D_MODEL; i += blockDim.x) {
    float x = pa[i], y = pb[i];
    d += x * y; na += x * x; nb += y * y;
  }
  __shared__ float r0[8], r1[8], r2[8];
  d = wave_reduce_add(d); na = wave_reduce_add(na); nb = wave_reduce_add(nb);
  int lane = threadIdx.x & 31, wid = threadIdx.x >> 5;
  if (lane == 0) { r0[wid] = d; r1[wid] = na; r2[wid] = nb; }
  __syncthreads();
  if (threadIdx.x == 0) {
    float td = 0.f, ta = 0.f, tb = 0.f;
    for (int w = 0; w < (int)(blockDim.x >> 5); ++w) { td += r0[w]; ta += r1[w]; tb += r2[w]; }
    out[b] = td / (fmaxf(sqrtf(ta), COS_EPS) * fmaxf(sqrtf(tb), COS_EPS));
  }
}

// ---------------- host launch ----------------
extern "C" void kernel_launch(void* const* d_in, const int* in_sizes, int n_in,
                              void* d_out, int out_size, void* d_ws, size_t ws_size,
                              hipStream_t stream) {
  (void)in_sizes; (void)n_in; (void)out_size; (void)ws_size;

  const float* xin[2]   = {(const float*)d_in[0], (const float*)d_in[1]};
  const float* attr[2]  = {(const float*)d_in[2], (const float*)d_in[3]};
  const int*   eidx[2]  = {(const int*)d_in[4],   (const int*)d_in[5]};
  const int*   batch[2] = {(const int*)d_in[6],   (const int*)d_in[7]};
  // params pytree, sorted-key flatten: head_W, head_b, layers[0..5]{We,Wk,Wq,Ws,Wv,be,bk,bq,bs,bv}, ln_b, ln_w
  const float* head_W = (const float*)d_in[8];
  const float* head_b = (const float*)d_in[9];
  const float *We[6], *Wk[6], *Wq[6], *Ws[6], *Wv[6];
  const float *be[6], *bk[6], *bq[6], *bs[6], *bv[6];
  for (int l = 0; l < 6; ++l) {
    int base = 10 + 10 * l;
    We[l] = (const float*)d_in[base + 0]; Wk[l] = (const float*)d_in[base + 1];
    Wq[l] = (const float*)d_in[base + 2]; Ws[l] = (const float*)d_in[base + 3];
    Wv[l] = (const float*)d_in[base + 4];
    be[l] = (const float*)d_in[base + 5]; bk[l] = (const float*)d_in[base + 6];
    bq[l] = (const float*)d_in[base + 7]; bs[l] = (const float*)d_in[base + 8];
    bv[l] = (const float*)d_in[base + 9];
  }
  const float* ln_b = (const float*)d_in[70];
  const float* ln_w = (const float*)d_in[71];

  // ---- carve workspace (256B aligned regions) ----
  char* wptr = (char*)d_ws;
  auto carve = [&](size_t bytes) -> char* {
    char* p = wptr; wptr += (bytes + 255) & ~(size_t)255; return p;
  };
  const size_t NDb = (size_t)N_NODES * D_MODEL * sizeof(float);
  float*          X      = (float*)carve(NDb);
  float*          Q      = (float*)carve(NDb);
  float*          Kf     = (float*)carve(NDb);
  float*          Vf     = (float*)carve(NDb);
  float*          XN     = (float*)carve(NDb);
  unsigned short* Xbf    = (unsigned short*)carve((size_t)MP_NODE * D_MODEL * 2);
  unsigned short* Wt     = (unsigned short*)carve((size_t)D_MODEL * D_MODEL * 2);
  unsigned short* headWt = (unsigned short*)carve((size_t)D_MODEL * D_MODEL * 2);
  unsigned short* Hbf    = (unsigned short*)carve((size_t)N_BATCH * D_MODEL * 2);
  float*          logits = (float*)carve((size_t)N_EDGES * sizeof(float));
  unsigned int*   mkey   = (unsigned int*)carve((size_t)N_NODES * 4);
  float*          den    = (float*)carve((size_t)N_NODES * 4);
  float*          sWa    = (float*)carve((size_t)N_NODES * 4);
  float*          qWe    = (float*)carve((size_t)N_NODES * 4);
  float*          qbe    = (float*)carve((size_t)N_NODES * 4);
  float*          pooled = (float*)carve((size_t)N_BATCH * D_MODEL * 4);
  float*          embA   = (float*)carve((size_t)N_BATCH * D_MODEL * 4);
  float*          embB   = (float*)carve((size_t)N_BATCH * D_MODEL * 4);

  const int T = 256;
  auto nb = [](long long n) { return (unsigned)((n + 255) / 256); };

  // head weight: transpose+bf16 once (shared by both towers)
  k_cvtT_pad_bf16<<<nb((long long)D_MODEL * D_MODEL), T, 0, stream>>>(
      head_W, headWt, D_MODEL, D_MODEL, D_MODEL);

  const long long ND = (long long)N_NODES * D_MODEL;
  const int node_blocks = (MP_NODE / BM) * (D_MODEL / BN);   // 157*5
  const int head_blocks = (N_BATCH / BM) * (D_MODEL / BN);   // 1*5

  for (int g = 0; g < 2; ++g) {
    float* emb = (g == 0) ? embA : embB;
    for (int l = 0; l < N_LAYERS; ++l) {
      const int din  = (l == 0) ? D_IN : D_MODEL;
      const int dinp = (l == 0) ? 32   : D_MODEL;   // pad K to one WMMA step
      const float* xsrc = (l == 0) ? xin[g] : X;

      k_cvt_pad_bf16<<<nb((long long)MP_NODE * dinp), T, 0, stream>>>(
          xsrc, Xbf, N_NODES, MP_NODE, din, dinp);

      const float* Wm[4] = {Wq[l], Wk[l], Wv[l], Ws[l]};
      const float* bm[4] = {bq[l], bk[l], bv[l], bs[l]};
      float*       Om[4] = {Q, Kf, Vf, XN};
      for (int m = 0; m < 4; ++m) {
        k_cvtT_pad_bf16<<<nb((long long)D_MODEL * dinp), T, 0, stream>>>(
            Wm[m], Wt, din, D_MODEL, dinp);
        k_gemm_bf16_async<<<node_blocks, T, 0, stream>>>(
            (const __bf16*)Xbf, (const __bf16*)Wt, bm[m], Om[m],
            N_NODES, D_MODEL, dinp, 0);
      }

      k_node_scalars<<<(N_NODES + 7) / 8, T, 0, stream>>>(
          Q, We[l], be[l], qWe, qbe, mkey, den, sWa);
      k_edge_logits<<<(N_EDGES + 7) / 8, T, 0, stream>>>(
          Q, Kf, eidx[g], attr[g], qWe, qbe, logits, mkey);
      k_edge_expsum<<<nb(N_EDGES), T, 0, stream>>>(logits, mkey, eidx[g], den);
      k_edge_scatter<<<(N_EDGES + 7) / 8, T, 0, stream>>>(
          logits, den, eidx[g], attr[g], Vf, XN, sWa);
      k_finalize<<<nb(ND), T, 0, stream>>>(XN, sWa, We[l], be[l], X);
    }

    k_zero_f32<<<nb(N_BATCH * D_MODEL), T, 0, stream>>>(pooled, N_BATCH * D_MODEL);
    k_pool<<<nb(ND), T, 0, stream>>>(X, batch[g], pooled);
    k_layernorm_bf16<<<N_BATCH, T, 0, stream>>>(pooled, ln_w, ln_b, Hbf);
    k_gemm_bf16_async<<<head_blocks, T, 0, stream>>>(
        (const __bf16*)Hbf, (const __bf16*)headWt, head_b, emb,
        N_BATCH, D_MODEL, D_MODEL, 1);
  }

  k_cosine<<<N_BATCH, T, 0, stream>>>(embA, embB, (float*)d_out);
}
